// CrossAttentionTransformer_21028159881867
// MI455X (gfx1250) — compile-verified, matrix operand order checked by reading, not run on hardware
//
#include <hip/hip_runtime.h>
#include <hip/hip_bf16.h>
#include <math.h>

// ---------------------------------------------------------------------------
// CDNA5 (gfx1250) wave32 WMMA types
// ---------------------------------------------------------------------------
typedef __attribute__((ext_vector_type(16))) _Float16 v16h;
typedef __attribute__((ext_vector_type(8)))  _Float16 v8h;
typedef __attribute__((ext_vector_type(2)))  _Float16 v2h;
typedef __attribute__((ext_vector_type(8)))  float    v8f;

// ---------------------------------------------------------------------------
// Batched WMMA GEMM:  C[z] = epilogue( alpha*A[z] (M x K) x B[z] (K x N) )
//
// - Workgroup: 256 threads = 8 waves, computes a 128(M) x 64(N) tile of C.
// - Per K-step (32): A-tile 128x32 and B-tile 64x32 (N-major) staged to LDS as
//   f16; each wave computes a 16x64 strip as 4x v_wmma_f32_16x16x32_f16,
//   reusing one A fragment (two ds_load_b128 per fragment).
// - Interior tiles (tile fully inside M/N): unguarded float2-vectorized
//   staging (global_load_b64 -> packed 2xf16 ds_store_b32); ragged K handled
//   by full-step loop + one k-guarded remainder step.  Edge tiles: clamped
//   branchless scalar staging (unconditional load + v_cndmask zero-fill).
// - LDS row stride 40 halfs (80B = 20 banks: conflict-free across the 16
//   fragment-owner lanes; 16B aligned -> fragments load as ds_load_b128).
// - Fragment K pattern (ISA 7.12.2): lane<16 -> K {0..7,16..23}, lane>=16 ->
//   K {8..15,24..31}; owner index = lane&15 (M for A, N for B).
// - C/D layout: VGPR r -> row (lane<16 ? r : 8+r), col = lane&15.
// - Two-level batching z = b*batchH + h (per-head views of [B, n, H*dh]).
// - Epilogue order: +bias[col], GELU(tanh), *rowscale[z*M+row], +resid.
// ---------------------------------------------------------------------------
template <int TA, int TB>
__global__ void wmma_gemm_f16(
    const float* __restrict__ A, const float* __restrict__ B,
    float* __restrict__ C,
    const float* __restrict__ bias,      // per-N, nullable
    const float* __restrict__ resid,     // same layout as C, nullable
    const float* __restrict__ rowscale,  // [z*M + row], nullable
    int M, int N, int K,
    int lda, int ldb, int ldc,
    long long sAb, long long sAh,
    long long sBb, long long sBh,
    long long sCb, long long sCh,
    int batchH, float alpha, int do_gelu)
{
    __shared__ __align__(16) _Float16 sA[128 * 40];
    __shared__ __align__(16) _Float16 sB[64 * 40];

    const int z  = blockIdx.z;
    const int bb = z / batchH;
    const int hh = z % batchH;
    const float* Ap = A + (long long)bb * sAb + (long long)hh * sAh;
    const float* Bp = B + (long long)bb * sBb + (long long)hh * sBh;
    float*       Cp = C + (long long)bb * sCb + (long long)hh * sCh;
    const float* Rp = resid ? resid + (long long)bb * sCb + (long long)hh * sCh : nullptr;

    const int tileM = blockIdx.y * 128;
    const int tileN = blockIdx.x * 64;
    const int tid   = threadIdx.x;
    const int lane  = tid & 31;
    const int wv    = tid >> 5;           // wave id 0..7 -> M rows wv*16..+15
    const int l16   = lane & 15;
    const int sel   = lane >> 4;          // 0: K {0..7,16..23}; 1: K {8..15,24..31}

    v8f acc[4] = {};                       // 4 column groups of 16

    // one K-step of WMMA compute out of staged LDS tiles
    auto mma_step = [&]() {
        __syncthreads();
        const int arow = wv * 16 + l16;
        const v8h a0 = *(const v8h*)&sA[arow * 40 + sel * 8];
        const v8h a1 = *(const v8h*)&sA[arow * 40 + 16 + sel * 8];
        const v16h af = __builtin_shufflevector(a0, a1,
            0, 1, 2, 3, 4, 5, 6, 7, 8, 9, 10, 11, 12, 13, 14, 15);
#pragma unroll
        for (int cg = 0; cg < 4; ++cg) {
            const int bcol = cg * 16 + l16;
            const v8h b0 = *(const v8h*)&sB[bcol * 40 + sel * 8];
            const v8h b1 = *(const v8h*)&sB[bcol * 40 + 16 + sel * 8];
            const v16h bf = __builtin_shufflevector(b0, b1,
                0, 1, 2, 3, 4, 5, 6, 7, 8, 9, 10, 11, 12, 13, 14, 15);
            acc[cg] = __builtin_amdgcn_wmma_f32_16x16x32_f16(
                false, af, false, bf, (short)0, acc[cg], false, false);
        }
        __syncthreads();
    };

    // branchless guarded scalar staging (clamped address + select-zero)
    auto stage_guarded = [&](int k0) {
#pragma unroll
        for (int t = 0; t < 16; ++t) {            // A: 128x32 = 4096 elements
            const int idx = tid + t * 256;
            int r, k;
            if (TA) { r = idx & 127; k = idx >> 7; }
            else    { k = idx & 31;  r = idx >> 5; }
            const int gi = tileM + r, gk = k0 + k;
            const int ci = gi < M ? gi : M - 1;
            const int ck = gk < K ? gk : K - 1;
            float v = TA ? Ap[(long long)ck * lda + ci]
                         : Ap[(long long)ci * lda + ck];
            v = (gi < M && gk < K) ? v : 0.0f;
            sA[r * 40 + k] = (_Float16)(v * alpha);
        }
#pragma unroll
        for (int t = 0; t < 8; ++t) {             // B: 64x32 = 2048 elements
            const int idx = tid + t * 256;
            int j, k;
            if (TB) { k = idx & 31; j = idx >> 5; }
            else    { j = idx & 63; k = idx >> 6; }
            const int gj = tileN + j, gk = k0 + k;
            const int cj = gj < N ? gj : N - 1;
            const int ck = gk < K ? gk : K - 1;
            float v = TB ? Bp[(long long)cj * ldb + gk < K ? (long long)cj * ldb + ck
                                                           : (long long)cj * ldb + ck]
                         : Bp[(long long)ck * ldb + cj];
            if (TB) v = Bp[(long long)cj * ldb + ck];
            v = (gj < N && gk < K) ? v : 0.0f;
            sB[j * 40 + k] = (_Float16)v;
        }
    };

    const bool interiorMN = (tileM + 128 <= M) && (tileN + 64 <= N);

    if (interiorMN) {
        const int Kfull = K & ~31;
        for (int k0 = 0; k0 < Kfull; k0 += 32) {
            // prefetch next K-step (one cacheline hint per thread)
            if (k0 + 32 < Kfull) {
                const float* pfA = TA
                    ? &Ap[(long long)(k0 + 32 + (tid & 31)) * lda + tileM]
                    : &Ap[(long long)(tileM + (tid & 127)) * lda + k0 + 32];
                __builtin_prefetch(pfA, 0, 0);
            }
            // ---- A: 2048 float2 loads (8 per thread), vectorized ----
            if (TA == 0) {
#pragma unroll
                for (int t = 0; t < 8; ++t) {
                    const int idx = tid + t * 256;
                    const int r = idx >> 4, k2 = (idx & 15) * 2;
                    const float2 v = *(const float2*)&Ap[(long long)(tileM + r) * lda + k0 + k2];
                    *(v2h*)&sA[r * 40 + k2] =
                        (v2h){(_Float16)(v.x * alpha), (_Float16)(v.y * alpha)};
                }
            } else {
#pragma unroll
                for (int t = 0; t < 8; ++t) {
                    const int idx = tid + t * 256;
                    const int r2 = (idx & 63) * 2, k = idx >> 6;
                    const float2 v = *(const float2*)&Ap[(long long)(k0 + k) * lda + tileM + r2];
                    sA[r2 * 40 + k]       = (_Float16)(v.x * alpha);
                    sA[(r2 + 1) * 40 + k] = (_Float16)(v.y * alpha);
                }
            }
            // ---- B: 1024 float2 loads (4 per thread) ----
            if (TB == 1) {
#pragma unroll
                for (int t = 0; t < 4; ++t) {
                    const int idx = tid + t * 256;
                    const int j = idx >> 4, k2 = (idx & 15) * 2;
                    const float2 v = *(const float2*)&Bp[(long long)(tileN + j) * ldb + k0 + k2];
                    *(v2h*)&sB[j * 40 + k2] = (v2h){(_Float16)v.x, (_Float16)v.y};
                }
            } else {
#pragma unroll
                for (int t = 0; t < 4; ++t) {
                    const int idx = tid + t * 256;
                    const int j2 = (idx & 31) * 2, k = idx >> 5;
                    const float2 v = *(const float2*)&Bp[(long long)(k0 + k) * ldb + tileN + j2];
                    sB[j2 * 40 + k]       = (_Float16)v.x;
                    sB[(j2 + 1) * 40 + k] = (_Float16)v.y;
                }
            }
            mma_step();
        }
        if (Kfull < K) {                  // ragged-K remainder (k-guard only)
            stage_guarded(Kfull);
            mma_step();
        }
    } else {
        for (int k0 = 0; k0 < K; k0 += 32) {
            stage_guarded(k0);
            mma_step();
        }
    }

    // ---- epilogue + store ----
    const int ibase = tileM + wv * 16 + (sel ? 8 : 0);
#pragma unroll
    for (int cg = 0; cg < 4; ++cg) {
        const int j = tileN + cg * 16 + l16;
#pragma unroll
        for (int r = 0; r < 8; ++r) {
            const int i = ibase + r;
            if (interiorMN || (i < M && j < N)) {
                float v = acc[cg][r];
                if (bias) v += bias[j];
                if (do_gelu) {
                    float t = 0.7978845608028654f * (v + 0.044715f * v * v * v);
                    v = 0.5f * v * (1.0f + tanhf(t));
                }
                if (rowscale) v *= rowscale[(long long)z * M + i];
                const long long off = (long long)i * ldc + j;
                if (Rp) v += Rp[off];
                Cp[off] = v;
            }
        }
    }
}

// ---------------------------------------------------------------------------
// LayerNorm over last dim D (<=256), one row per block of 256 threads
// ---------------------------------------------------------------------------
__global__ void ln_kernel(const float* __restrict__ x, const float* __restrict__ w,
                          const float* __restrict__ b, float* __restrict__ y, int D)
{
    const long long row = blockIdx.x;
    const float* xr = x + row * (long long)D;
    float*       yr = y + row * (long long)D;
    __shared__ float sred[256];
    const int tid = threadIdx.x;
    const float v = (tid < D) ? xr[tid] : 0.0f;
    sred[tid] = v; __syncthreads();
    for (int s = 128; s > 0; s >>= 1) { if (tid < s) sred[tid] += sred[tid + s]; __syncthreads(); }
    const float mu = sred[0] / (float)D;
    __syncthreads();
    const float d = (tid < D) ? (v - mu) : 0.0f;
    sred[tid] = d * d; __syncthreads();
    for (int s = 128; s > 0; s >>= 1) { if (tid < s) sred[tid] += sred[tid + s]; __syncthreads(); }
    const float var = sred[0] / (float)D;
    if (tid < D) yr[tid] = d * rsqrtf(var + 1e-5f) * w[tid] + b[tid];
}

// diag[b,h,i] = 0.5*dn^2 * sum_d q[b,i,h*dh+d]^2   (head view of [B,n,inner])
__global__ void diag_kernel(const float* __restrict__ q, float* __restrict__ diag,
                            int n, int dh, int inner, int Hh, float coef)
{
    const long long r = blockIdx.x;          // r = (b*H+h)*n + i
    const int i = (int)(r % n);
    const long long z = r / n;
    const int b = (int)(z / Hh), h = (int)(z % Hh);
    const float* qr = q + (long long)b * n * inner + (long long)h * dh + (long long)i * inner;
    __shared__ float sred[128];
    const int tid = threadIdx.x;
    const float v = (tid < dh) ? qr[tid] : 0.0f;
    sred[tid] = v * v; __syncthreads();
    for (int s = 64; s > 0; s >>= 1) { if (tid < s) sred[tid] += sred[tid + s]; __syncthreads(); }
    if (tid == 0) diag[r] = 0.5f * coef * sred[0];
}

// per-row max over Mf columns
__global__ void rowmax_kernel(const float* __restrict__ dd, float* __restrict__ out, int Mf)
{
    const long long r = blockIdx.x;
    const float* p = dd + r * (long long)Mf;
    __shared__ float sred[256];
    const int tid = threadIdx.x;
    float m = -3.4e38f;
    for (int j = tid; j < Mf; j += 256) m = fmaxf(m, p[j]);
    sred[tid] = m; __syncthreads();
    for (int s = 128; s > 0; s >>= 1) { if (tid < s) sred[tid] = fmaxf(sred[tid], sred[tid + s]); __syncthreads(); }
    if (tid == 0) out[r] = sred[0];
}

__global__ void gmax_partial_kernel(const float* __restrict__ x, float* __restrict__ part, long long total)
{
    __shared__ float sred[256];
    const int tid = threadIdx.x;
    float m = -3.4e38f;
    for (long long i = blockIdx.x * 256LL + tid; i < total; i += (long long)gridDim.x * 256LL)
        m = fmaxf(m, x[i]);
    sred[tid] = m; __syncthreads();
    for (int s = 128; s > 0; s >>= 1) { if (tid < s) sred[tid] = fmaxf(sred[tid], sred[tid + s]); __syncthreads(); }
    if (tid == 0) part[blockIdx.x] = sred[0];
}

__global__ void gmax_final_kernel(const float* __restrict__ part, int cnt, float* __restrict__ out)
{
    __shared__ float sred[256];
    const int tid = threadIdx.x;
    float m = -3.4e38f;
    for (int i = tid; i < cnt; i += 256) m = fmaxf(m, part[i]);
    sred[tid] = m; __syncthreads();
    for (int s = 128; s > 0; s >>= 1) { if (tid < s) sred[tid] = fmaxf(sred[tid], sred[tid + s]); __syncthreads(); }
    if (tid == 0) out[0] = sred[0];
}

// out = ratio * (exp(dd - diag[row] - stab) + eps); stab per-row (query) or scalar (key)
__global__ void favor_kernel(const float* __restrict__ dd, const float* __restrict__ diag,
                             const float* __restrict__ stab_row, const float* __restrict__ stab_scalar,
                             float* __restrict__ out, long long total, int Mf, float ratio, float eps)
{
    for (long long idx = blockIdx.x * 256LL + threadIdx.x; idx < total;
         idx += (long long)gridDim.x * 256LL) {
        const long long row = idx / Mf;
        const float st = stab_row ? stab_row[row] : stab_scalar[0];
        out[idx] = ratio * (expf(dd[idx] - diag[row] - st) + eps);
    }
}

// ksum[z,m] = sum_i kp[z,i,m]
__global__ void colsum_kernel(const float* __restrict__ kp, float* __restrict__ ksum,
                              int n, int Mf, long long total)
{
    for (long long idx = blockIdx.x * 256LL + threadIdx.x; idx < total;
         idx += (long long)gridDim.x * 256LL) {
        const long long z = idx / Mf;
        const int m = (int)(idx % Mf);
        const float* p = kp + z * (long long)n * Mf + m;
        float a = 0.0f;
        for (int i = 0; i < n; ++i) a += p[(long long)i * Mf];
        ksum[idx] = a;
    }
}

// dinv[z,i] = 1 / sum_m qp[z,i,m]*ksum[z,m]
__global__ void dinv_kernel(const float* __restrict__ qp, const float* __restrict__ ksum,
                            float* __restrict__ dinv, int n, int Mf)
{
    const long long r = blockIdx.x;
    const long long z = r / n;
    const float* q  = qp   + r * (long long)Mf;
    const float* ks = ksum + z * (long long)Mf;
    __shared__ float sred[256];
    const int tid = threadIdx.x;
    float a = 0.0f;
    for (int j = tid; j < Mf; j += 256) a += q[j] * ks[j];
    sred[tid] = a; __syncthreads();
    for (int s = 128; s > 0; s >>= 1) { if (tid < s) sred[tid] += sred[tid + s]; __syncthreads(); }
    if (tid == 0) dinv[r] = 1.0f / sred[0];
}

// X[b,f,s] = lin_b[f] + sum_c x[b,s,f,c]*lin_w[f,c] + pe(F=128 rows, d=256)[f,s]
__global__ void lin_pe_kernel(const float* __restrict__ x, const float* __restrict__ lw,
                              const float* __restrict__ lb, float* __restrict__ X, int Bc)
{
    const long long idx = blockIdx.x * 256LL + threadIdx.x;
    const long long total = (long long)Bc * 128 * 256;
    if (idx >= total) return;
    const int s = (int)(idx % 256);
    const long long t = idx / 256;
    const int f = (int)(t % 128);
    const int b = (int)(t / 128);
    const float* xp = x + ((((long long)b * 256 + s) * 128) + f) * 16;
    const float* wp = lw + f * 16;
    float a = lb[f];
#pragma unroll
    for (int c = 0; c < 16; ++c) a += xp[c] * wp[c];
    const int i2 = s & ~1;
    const float div = expf((float)i2 * (-9.210340371976184f / 256.0f));
    const float ang = (float)f * div;
    a += (s & 1) ? cosf(ang) : sinf(ang);
    X[idx] = a;   // [b][f][s]
}

// Y[b,s,f] = X[b,f,s] + pe(S=256 rows, d=128)[s,f]
__global__ void transpose_pe_kernel(const float* __restrict__ X, float* __restrict__ Y, int Bc)
{
    const long long idx = blockIdx.x * 256LL + threadIdx.x;
    const long long total = (long long)Bc * 256 * 128;
    if (idx >= total) return;
    const int f = (int)(idx % 128);
    const long long t = idx / 128;
    const int s = (int)(t % 256);
    const int b = (int)(t / 256);
    float v = X[((long long)b * 128 + f) * 256 + s];
    const int i2 = f & ~1;
    const float div = expf((float)i2 * (-9.210340371976184f / 128.0f));
    const float ang = (float)s * div;
    v += (f & 1) ? cosf(ang) : sinf(ang);
    Y[idx] = v;   // [b][s][f]
}

// out[b,f] = mean_s Y[b,s,f]
__global__ void mean_kernel(const float* __restrict__ Y, float* __restrict__ out, int Bc)
{
    const int idx = blockIdx.x * 256 + threadIdx.x;
    if (idx >= Bc * 128) return;
    const int f = idx % 128, b = idx / 128;
    const float* p = Y + (long long)b * 256 * 128 + f;
    float a = 0.0f;
    for (int s = 0; s < 256; ++s) a += p[(long long)s * 128];
    out[idx] = a * (1.0f / 256.0f);
}

// ---------------------------------------------------------------------------
// Host orchestration
// ---------------------------------------------------------------------------
namespace {

struct Perf {
    int Bc, n, D, H, dh, inner, Mf;
    float dn, ratio;
    const float *ln1w, *ln1b, *wq, *wk, *wv, *wo, *wob;
    const float *ln2w, *ln2b, *f1w, *f1b, *f2w, *f2b, *proj;
};

struct Ws {
    float *X, *X2, *Hb, *Q, *Kb, *V, *DDQ, *DDK;
    float *DIAGQ, *DIAGK, *STABQ, *PART, *KSUM, *DINV, *CTX, *ATTN, *FF1;
};

static inline void launch_gemm(hipStream_t st,
    const float* A, const float* B, float* C,
    const float* bias, const float* resid, const float* rowscale,
    int M, int N, int K, int lda, int ldb, int ldc,
    long long sAb, long long sAh, long long sBb, long long sBh,
    long long sCb, long long sCh,
    int batchB, int batchH, int transA, int transB, float alpha, int gelu)
{
    dim3 grid((N + 63) / 64, (M + 127) / 128, batchB * batchH);
    if (!transA && !transB)
        wmma_gemm_f16<0, 0><<<grid, 256, 0, st>>>(A, B, C, bias, resid, rowscale,
            M, N, K, lda, ldb, ldc, sAb, sAh, sBb, sBh, sCb, sCh, batchH, alpha, gelu);
    else if (!transA && transB)
        wmma_gemm_f16<0, 1><<<grid, 256, 0, st>>>(A, B, C, bias, resid, rowscale,
            M, N, K, lda, ldb, ldc, sAb, sAh, sBb, sBh, sCb, sCh, batchH, alpha, gelu);
    else
        wmma_gemm_f16<1, 0><<<grid, 256, 0, st>>>(A, B, C, bias, resid, rowscale,
            M, N, K, lda, ldb, ldc, sAb, sAh, sBb, sBh, sCb, sCh, batchH, alpha, gelu);
}

static void run_performer(hipStream_t st, const Perf& p, float* X, const Ws& w)
{
    const int Bc = p.Bc, n = p.n, D = p.D, H = p.H, dh = p.dh, I = p.inner, Mf = p.Mf;
    const long long rowsBH = (long long)Bc * H * n;
    const long long totDD  = rowsBH * (long long)Mf;
    const long long totKS  = (long long)Bc * H * Mf;

    for (int l = 0; l < 4; ++l) {
        const float* ln1w = p.ln1w + (long long)l * D;
        const float* ln1b = p.ln1b + (long long)l * D;
        const float* wq   = p.wq   + (long long)l * D * I;
        const float* wk   = p.wk   + (long long)l * D * I;
        const float* wv   = p.wv   + (long long)l * D * I;
        const float* wo   = p.wo   + (long long)l * I * D;
        const float* wob  = p.wob  + (long long)l * D;
        const float* ln2w = p.ln2w + (long long)l * D;
        const float* ln2b = p.ln2b + (long long)l * D;
        const float* f1w  = p.f1w  + (long long)l * D * 4 * D;
        const float* f1b  = p.f1b  + (long long)l * 4 * D;
        const float* f2w  = p.f2w  + (long long)l * 4 * D * D;
        const float* f2b  = p.f2b  + (long long)l * D;
        const float* proj = p.proj + (long long)l * Mf * dh;

        // h = LN1(x)
        ln_kernel<<<Bc * n, 256, 0, st>>>(X, ln1w, ln1b, w.Hb, D);

        // Q/K/V = h @ w{q,k,v}   [B, n, I]
        launch_gemm(st, w.Hb, wq, w.Q,  nullptr, nullptr, nullptr, n, I, D, D, I, I,
                    (long long)n * D, 0, 0, 0, (long long)n * I, 0, Bc, 1, 0, 0, 1.0f, 0);
        launch_gemm(st, w.Hb, wk, w.Kb, nullptr, nullptr, nullptr, n, I, D, D, I, I,
                    (long long)n * D, 0, 0, 0, (long long)n * I, 0, Bc, 1, 0, 0, 1.0f, 0);
        launch_gemm(st, w.Hb, wv, w.V,  nullptr, nullptr, nullptr, n, I, D, D, I, I,
                    (long long)n * D, 0, 0, 0, (long long)n * I, 0, Bc, 1, 0, 0, 1.0f, 0);

        // dd = (q*dn) @ proj^T    [B*H, n, Mf]
        launch_gemm(st, w.Q, proj, w.DDQ, nullptr, nullptr, nullptr, n, Mf, dh, I, dh, Mf,
                    (long long)n * I, dh, 0, 0,
                    (long long)H * n * Mf, (long long)n * Mf, Bc, H, 0, 1, p.dn, 0);
        launch_gemm(st, w.Kb, proj, w.DDK, nullptr, nullptr, nullptr, n, Mf, dh, I, dh, Mf,
                    (long long)n * I, dh, 0, 0,
                    (long long)H * n * Mf, (long long)n * Mf, Bc, H, 0, 1, p.dn, 0);

        // diag = 0.5*dn^2*||q||^2 per row
        diag_kernel<<<(unsigned)rowsBH, 128, 0, st>>>(w.Q,  w.DIAGQ, n, dh, I, H, p.dn * p.dn);
        diag_kernel<<<(unsigned)rowsBH, 128, 0, st>>>(w.Kb, w.DIAGK, n, dh, I, H, p.dn * p.dn);

        // stab: per-row max (query), global max (key)
        rowmax_kernel<<<(unsigned)rowsBH, 256, 0, st>>>(w.DDQ, w.STABQ, Mf);
        gmax_partial_kernel<<<1024, 256, 0, st>>>(w.DDK, w.PART, totDD);
        gmax_final_kernel<<<1, 256, 0, st>>>(w.PART, 1024, w.PART + 1024);

        // qp/kp = ratio*(exp(dd - diag - stab) + eps)   (in place)
        favor_kernel<<<4096, 256, 0, st>>>(w.DDQ, w.DIAGQ, w.STABQ, nullptr,
                                           w.DDQ, totDD, Mf, p.ratio, 1e-4f);
        favor_kernel<<<4096, 256, 0, st>>>(w.DDK, w.DIAGK, nullptr, w.PART + 1024,
                                           w.DDK, totDD, Mf, p.ratio, 1e-4f);

        // ksum, d_inv
        colsum_kernel<<<(unsigned)((totKS + 255) / 256), 256, 0, st>>>(w.DDK, w.KSUM, n, Mf, totKS);
        dinv_kernel<<<(unsigned)rowsBH, 256, 0, st>>>(w.DDQ, w.KSUM, w.DINV, n, Mf);

        // ctx = kp^T @ v    [B*H, Mf, dh]
        launch_gemm(st, w.DDK, w.V, w.CTX, nullptr, nullptr, nullptr, Mf, dh, n,
                    Mf, I, dh,
                    (long long)H * n * Mf, (long long)n * Mf,
                    (long long)n * I, dh,
                    (long long)H * Mf * dh, (long long)Mf * dh,
                    Bc, H, 1, 0, 1.0f, 0);

        // attn = (qp @ ctx) * d_inv  -> per-head slices of [B, n, I]
        launch_gemm(st, w.DDQ, w.CTX, w.ATTN, nullptr, nullptr, w.DINV, n, dh, Mf,
                    Mf, dh, I,
                    (long long)H * n * Mf, (long long)n * Mf,
                    (long long)H * Mf * dh, (long long)Mf * dh,
                    (long long)n * I, dh,
                    Bc, H, 0, 0, 1.0f, 0);

        // x = x + attn @ wo + wo_b
        launch_gemm(st, w.ATTN, wo, X, wob, X, nullptr, n, D, I, I, D, D,
                    (long long)n * I, 0, 0, 0, (long long)n * D, 0, Bc, 1, 0, 0, 1.0f, 0);

        // FFN: h2 = LN2(x); x = x + gelu(h2@f1w+f1b)@f2w + f2b
        ln_kernel<<<Bc * n, 256, 0, st>>>(X, ln2w, ln2b, w.Hb, D);
        launch_gemm(st, w.Hb, f1w, w.FF1, f1b, nullptr, nullptr, n, 4 * D, D, D, 4 * D, 4 * D,
                    (long long)n * D, 0, 0, 0, (long long)n * 4 * D, 0, Bc, 1, 0, 0, 1.0f, 1);
        launch_gemm(st, w.FF1, f2w, X, f2b, X, nullptr, n, D, 4 * D, 4 * D, D, D,
                    (long long)n * 4 * D, 0, 0, 0, (long long)n * D, 0, Bc, 1, 0, 0, 1.0f, 0);
    }
}

} // namespace

extern "C" void kernel_launch(void* const* d_in, const int* in_sizes, int n_in,
                              void* d_out, int out_size, void* d_ws, size_t ws_size,
                              hipStream_t stream)
{
    (void)in_sizes; (void)n_in; (void)out_size; (void)ws_size;
    const int B = 64;

    const float* x_in  = (const float*)d_in[0];
    const float* lin_w = (const float*)d_in[1];
    const float* lin_b = (const float*)d_in[2];
    const float* t[14]; for (int i = 0; i < 14; ++i) t[i] = (const float*)d_in[3 + i];
    const float* m[14]; for (int i = 0; i < 14; ++i) m[i] = (const float*)d_in[17 + i];

    // ---- workspace carve (f32 elements); requires ~450 MB ----
    float* ws = (float*)d_ws;
    size_t off = 0;
    auto carve = [&](size_t cnt) { float* p = ws + off; off += cnt; return p; };
    Ws w;
    w.X     = carve(2097152);   // [64,128,256] ticker-major
    w.X2    = carve(2097152);   // [64,256,128] time-major
    w.Hb    = carve(2097152);   // LN output
    w.Q     = carve(5242880);   // [64,128,640] max
    w.Kb    = carve(5242880);
    w.V     = carve(5242880);
    w.DDQ   = carve(25395200);  // [320,128,620] max
    w.DDK   = carve(25395200);
    w.DIAGQ = carve(65536);
    w.DIAGK = carve(65536);
    w.STABQ = carve(65536);
    w.PART  = carve(1056);      // 1024 partials + scalar at [1024]
    w.KSUM  = carve(198400);
    w.DINV  = carve(65536);
    w.CTX   = carve(25395200);  // [320,620,128] max
    w.ATTN  = carve(5242880);
    w.FF1   = carve(8388608);   // [64,128,1024] / [64,256,512]

    // ---- stage 0: per-file linear + ticker-major transpose + pos-enc ----
    {
        const long long total = (long long)B * 128 * 256;
        lin_pe_kernel<<<(unsigned)((total + 255) / 256), 256, 0, stream>>>(
            x_in, lin_w, lin_b, w.X, B);
    }

    // ---- ticker-major performer: n=128, D=256, 5 heads, dh=128, m=620 ----
    Perf tp;
    tp.Bc = B; tp.n = 128; tp.D = 256; tp.H = 5; tp.dh = 128; tp.inner = 640; tp.Mf = 620;
    tp.dn = powf(128.0f, -0.25f); tp.ratio = 1.0f / sqrtf(620.0f);
    tp.ln1w = t[0]; tp.ln1b = t[1]; tp.wq = t[2]; tp.wk = t[3]; tp.wv = t[4];
    tp.wo = t[5]; tp.wob = t[6]; tp.ln2w = t[7]; tp.ln2b = t[8];
    tp.f1w = t[9]; tp.f1b = t[10]; tp.f2w = t[11]; tp.f2b = t[12]; tp.proj = t[13];
    run_performer(stream, tp, w.X, w);

    // ---- transpose to time-major + pos-enc ----
    {
        const long long total = (long long)B * 256 * 128;
        transpose_pe_kernel<<<(unsigned)((total + 255) / 256), 256, 0, stream>>>(w.X, w.X2, B);
    }

    // ---- time-major performer: n=256, D=128, 4 heads, dh=64, m=266 ----
    Perf mp;
    mp.Bc = B; mp.n = 256; mp.D = 128; mp.H = 4; mp.dh = 64; mp.inner = 256; mp.Mf = 266;
    mp.dn = powf(64.0f, -0.25f); mp.ratio = 1.0f / sqrtf(266.0f);
    mp.ln1w = m[0]; mp.ln1b = m[1]; mp.wq = m[2]; mp.wk = m[3]; mp.wv = m[4];
    mp.wo = m[5]; mp.wob = m[6]; mp.ln2w = m[7]; mp.ln2b = m[8];
    mp.f1w = m[9]; mp.f1b = m[10]; mp.f2w = m[11]; mp.f2b = m[12]; mp.proj = m[13];
    run_performer(stream, mp, w.X2, w);

    // ---- final mean over sequence -> [B, F] ----
    mean_kernel<<<(B * 128 + 255) / 256, 256, 0, stream>>>(w.X2, (float*)d_out, B);
}
